// GATmGCN_44487271252597
// MI455X (gfx1250) — matrix operand order repaired and checked by reading, not compiled, not run on hardware
//
#include <hip/hip_runtime.h>
#include <hip/hip_bf16.h>

// ---------------------------------------------------------------------------
// Types for CDNA5 WMMA (wave32): v_wmma_f32_16x16x32_bf16
// ---------------------------------------------------------------------------
typedef __attribute__((ext_vector_type(16))) __bf16 v16bf;
typedef __attribute__((ext_vector_type(8)))  float  v8f;

union AFrag { v16bf v; unsigned short u[16]; uint4 q[2]; };
union CFrag { v8f   v; float          f[8];  };

#define LRELU_ALPHA 0.2f

// ---------------------------------------------------------------------------
// helpers
// ---------------------------------------------------------------------------
__device__ __forceinline__ unsigned short f2bf(float f) {
    unsigned int b = __float_as_uint(f);
    unsigned int r = (b + 0x7FFFu + ((b >> 16) & 1u)) >> 16;
    return (unsigned short)r;
}
__device__ __forceinline__ unsigned int enc_f32(float f) {
    unsigned int b = __float_as_uint(f);
    return (b & 0x80000000u) ? ~b : (b | 0x80000000u);
}
__device__ __forceinline__ float dec_f32(unsigned int u) {
    return (u & 0x80000000u) ? __uint_as_float(u & 0x7FFFFFFFu)
                             : __uint_as_float(~u);
}

// ---------------------------------------------------------------------------
// fill / convert kernels
// ---------------------------------------------------------------------------
__global__ void k_zero_f32(float* p, long long n) {
    long long i = (long long)blockIdx.x * blockDim.x + threadIdx.x;
    if (i < n) p[i] = 0.0f;
}
__global__ void k_fill_u32(unsigned int* p, long long n, unsigned int v) {
    long long i = (long long)blockIdx.x * blockDim.x + threadIdx.x;
    if (i < n) p[i] = v;
}
__global__ void k_f32_to_bf16(const float* __restrict__ in,
                              unsigned short* __restrict__ out, long long n) {
    long long i = (long long)blockIdx.x * blockDim.x + threadIdx.x;
    if (i < n) out[i] = f2bf(in[i]);
}
// gat_W0 [H=8][128][16] -> BT [128 cols][128 k]: BT[h*16+j][k] = W0[h][k][j]
__global__ void k_w0cat_T(const float* __restrict__ W0,
                          unsigned short* __restrict__ out) {
    int i = blockIdx.x * blockDim.x + threadIdx.x;   // i = c*128 + k
    if (i >= 128 * 128) return;
    int c = i >> 7, k = i & 127;
    int h = c >> 4, j = c & 15;
    out[i] = f2bf(W0[(h * 128 + k) * 16 + j]);
}
// gat_W1 [H=8][128][128] -> BT [1024 cols][128 k]: BT[h*128+j][k] = W1[h][k][j]
__global__ void k_w1cat_T(const float* __restrict__ W1,
                          unsigned short* __restrict__ out) {
    int i = blockIdx.x * blockDim.x + threadIdx.x;   // i = c*128 + k
    if (i >= 1024 * 128) return;
    int c = i >> 7, k = i & 127;
    int h = c >> 7, j = c & 127;
    out[i] = f2bf(W1[(h * 128 + k) * 128 + j]);
}
// generic 128x128 f32 -> transposed bf16: out[c*128 + k] = in[k*128 + c]
__global__ void k_transpose_bf16(const float* __restrict__ in,
                                 unsigned short* __restrict__ out) {
    int i = blockIdx.x * blockDim.x + threadIdx.x;
    if (i >= 128 * 128) return;
    int c = i >> 7, k = i & 127;
    out[i] = f2bf(in[k * 128 + c]);
}

// ---------------------------------------------------------------------------
// WMMA GEMM, specialized for this workload:
//   C[M x 128](f32) = A[M x 128](bf16, row-major) * BT[128 x 128](bf16)^T
// BT is column-major-of-B, i.e. BT[col][k]; all leading dims = 128.
// Block = 256 threads = 8 waves; block tile 128x64; wave tile 32x32
// (2x2 of 16x16x32 WMMA, K fully unrolled -> 16 WMMA/wave).
// No LDS: A fragments are contiguous b128 in row-major A; B fragments are
// contiguous b128 in BT; B is tiny and L2/WGP$-resident.
// ---------------------------------------------------------------------------
__global__ __launch_bounds__(256)
void k_gemm_bf16(const unsigned short* __restrict__ A,
                 const unsigned short* __restrict__ BT,
                 float* __restrict__ C, int M) {
    const int tid  = threadIdx.x;
    const int wave = tid >> 5;
    const int lane = tid & 31;
    const int lm   = lane & 15;      // row/col within 16
    const int lh   = lane >> 4;      // lane half: K-range / M-half select
    const int wm   = wave >> 1;      // 0..3 -> 32-row slice
    const int wn   = wave & 1;       // 0..1 -> 32-col slice

    const int rowBase = blockIdx.x * 128;
    const int colBase = blockIdx.y * 64;

    // per-lane A row pointers (clamped; clamped rows are never stored)
    const unsigned short* ap[2];
#pragma unroll
    for (int sm = 0; sm < 2; ++sm) {
        int r = rowBase + wm * 32 + sm * 16 + lm;
        if (r >= M) r = M - 1;
        ap[sm] = A + (size_t)r * 128;
    }
    // per-lane B column pointers
    const unsigned short* bp[2];
#pragma unroll
    for (int sn = 0; sn < 2; ++sn) {
        int c = colBase + wn * 32 + sn * 16 + lm;
        bp[sn] = BT + (size_t)c * 128;
    }

    v8f acc[2][2];
#pragma unroll
    for (int i = 0; i < 2; ++i)
#pragma unroll
        for (int j = 0; j < 2; ++j) acc[i][j] = 0.0f;

#pragma unroll
    for (int kk = 0; kk < 128; kk += 32) {
        AFrag af[2], bf[2];
        // A frag (16x32 MxK): u[0..7] = k (kk + lh*8 ..+7),
        //                     u[8..15] = k (kk+16+lh*8 ..+7)
#pragma unroll
        for (int sm = 0; sm < 2; ++sm) {
            af[sm].q[0] = *(const uint4*)(ap[sm] + kk + lh * 8);
            af[sm].q[1] = *(const uint4*)(ap[sm] + kk + 16 + lh * 8);
        }
        // B frag (32x16 KxN): u[0..7] = k (kk + lh*8 ..+7), u[8..15] = k (kk+16+lh*8 ..+7)
#pragma unroll
        for (int sn = 0; sn < 2; ++sn) {
            bf[sn].q[0] = *(const uint4*)(bp[sn] + kk + lh * 8);
            bf[sn].q[1] = *(const uint4*)(bp[sn] + kk + 16 + lh * 8);
        }
#pragma unroll
        for (int sm = 0; sm < 2; ++sm)
#pragma unroll
            for (int sn = 0; sn < 2; ++sn)
                acc[sm][sn] = __builtin_amdgcn_wmma_f32_16x16x32_bf16(
                    false, af[sm].v, false, bf[sn].v,
                    (short)0, acc[sm][sn], false, false);
    }

    // store: VGPR r -> M = m0 + r (m0 includes lh*8); N = colBase+... + lane%16
    const bool full = (rowBase + 128 <= M);
#pragma unroll
    for (int sm = 0; sm < 2; ++sm) {
#pragma unroll
        for (int sn = 0; sn < 2; ++sn) {
            CFrag cf; cf.v = acc[sm][sn];
            const int m0   = rowBase + wm * 32 + sm * 16 + lh * 8;
            const int ncol = colBase + wn * 32 + sn * 16 + lm;
            float* cp = C + (size_t)m0 * 128 + ncol;
            if (full) {
#pragma unroll
                for (int r = 0; r < 8; ++r) cp[r * 128] = cf.f[r];
            } else {
#pragma unroll
                for (int r = 0; r < 8; ++r)
                    if (m0 + r < M) cp[r * 128] = cf.f[r];
            }
        }
    }
}

// ---------------------------------------------------------------------------
// es/ed: per-node, per-head attention logit halves
// Wh row layout: [node][h*fp + j]; es/ed layout: [node*nh + h]
// ---------------------------------------------------------------------------
__global__ void k_esed(const float* __restrict__ Wh,
                       const float* __restrict__ a_s,
                       const float* __restrict__ a_d,
                       float* __restrict__ es, float* __restrict__ ed,
                       int n, int nh, int fp) {
    int idx = blockIdx.x * blockDim.x + threadIdx.x;
    if (idx >= n * nh) return;
    int node = idx / nh, h = idx % nh;
    const float* w  = Wh + (long long)node * nh * fp + h * fp;
    const float* as = a_s + h * fp;
    const float* ad = a_d + h * fp;
    float s = 0.f, d = 0.f;
    for (int j = 0; j < fp; ++j) { float v = w[j]; s += v * as[j]; d += v * ad[j]; }
    es[idx] = s; ed[idx] = d;
}

// ---------------------------------------------------------------------------
// edge pass 1: segment max of leaky_relu(es[src]+ed[dst]) per (dst, head)
// ---------------------------------------------------------------------------
__global__ void k_edge_max(const long long* __restrict__ src,
                           const long long* __restrict__ dst,
                           const float* __restrict__ es,
                           const float* __restrict__ ed,
                           unsigned int* __restrict__ mbuf,
                           long long E, int nh) {
    long long idx = (long long)blockIdx.x * blockDim.x + threadIdx.x;
    if (idx >= E * nh) return;
    long long e = idx / nh; int h = (int)(idx % nh);
    long long s = src[e], d = dst[e];
    float l = es[s * nh + h] + ed[d * nh + h];
    l = l > 0.f ? l : LRELU_ALPHA * l;
    atomicMax(&mbuf[d * nh + h], enc_f32(l));
}

// ---------------------------------------------------------------------------
// edge pass 2: z = exp(l - max); store z, accumulate segment sum
// ---------------------------------------------------------------------------
__global__ void k_edge_z(const long long* __restrict__ src,
                         const long long* __restrict__ dst,
                         const float* __restrict__ es,
                         const float* __restrict__ ed,
                         const unsigned int* __restrict__ mbuf,
                         float* __restrict__ zbuf,
                         float* __restrict__ sbuf,
                         long long E, int nh) {
    long long idx = (long long)blockIdx.x * blockDim.x + threadIdx.x;
    if (idx >= E * nh) return;
    long long e = idx / nh; int h = (int)(idx % nh);
    long long s = src[e], d = dst[e];
    float l = es[s * nh + h] + ed[d * nh + h];
    l = l > 0.f ? l : LRELU_ALPHA * l;
    float z = __expf(l - dec_f32(mbuf[d * nh + h]));
    zbuf[idx] = z;
    atomicAdd(&sbuf[d * nh + h], z);
}

// ---------------------------------------------------------------------------
// edge pass 3: att = z / segment-sum   (in place on zbuf)
// ---------------------------------------------------------------------------
__global__ void k_att(const long long* __restrict__ dst,
                      const float* __restrict__ sbuf,
                      float* __restrict__ zbuf, long long E, int nh) {
    long long idx = (long long)blockIdx.x * blockDim.x + threadIdx.x;
    if (idx >= E * nh) return;
    long long e = idx / nh; int h = (int)(idx % nh);
    zbuf[idx] /= fmaxf(sbuf[dst[e] * nh + h], 1e-16f);
}

// ---------------------------------------------------------------------------
// edge pass 4: out[dst, f] += att * Wh[src, f]   (one wave per edge)
// Ftot = nh*fp = 128; out row stride fixed 128
// ---------------------------------------------------------------------------
__global__ __launch_bounds__(256)
void k_gat_scatter(const long long* __restrict__ src,
                   const long long* __restrict__ dst,
                   const float* __restrict__ Wh,
                   const float* __restrict__ att,
                   float* __restrict__ out,
                   long long E, int nh, int fp) {
    long long wid = (long long)blockIdx.x * (blockDim.x >> 5) + (threadIdx.x >> 5);
    if (wid >= E) return;
    const int lane = threadIdx.x & 31;
    const long long s = src[wid], d = dst[wid];
    const int Ftot = nh * fp;
    for (int f = lane; f < Ftot; f += 32) {
        float a = att[wid * nh + f / fp];
        atomicAdd(&out[d * 128 + f], a * Wh[s * (long long)Ftot + f]);
    }
}

// ---------------------------------------------------------------------------
// GCN: degree accumulation + normalized scatter
// ---------------------------------------------------------------------------
__global__ void k_deg(const long long* __restrict__ dst,
                      const float* __restrict__ w,
                      float* __restrict__ deg, long long E) {
    long long e = (long long)blockIdx.x * blockDim.x + threadIdx.x;
    if (e < E) atomicAdd(&deg[dst[e]], w[e]);
}
__global__ __launch_bounds__(256)
void k_gcn_scatter(const long long* __restrict__ src,
                   const long long* __restrict__ dst,
                   const float* __restrict__ w,
                   const float* __restrict__ deg,
                   const float* __restrict__ hmat,
                   float* __restrict__ out, long long E) {
    long long wid = (long long)blockIdx.x * (blockDim.x >> 5) + (threadIdx.x >> 5);
    if (wid >= E) return;
    const int lane = threadIdx.x & 31;
    const long long s = src[wid], d = dst[wid];
    const float norm = w[wid] / fmaxf(deg[d], 1e-16f);
    for (int f = lane; f < 128; f += 32)
        atomicAdd(&out[d * 128 + f], norm * hmat[s * 128 + f]);
}

// ---------------------------------------------------------------------------
// activations + bf16 recompress, final combine
// ---------------------------------------------------------------------------
__global__ void k_elu_bf16(const float* __restrict__ in,
                           unsigned short* __restrict__ out, long long n) {
    long long i = (long long)blockIdx.x * blockDim.x + threadIdx.x;
    if (i >= n) return;
    float v = in[i];
    v = v > 0.f ? v : (__expf(v) - 1.f);
    out[i] = f2bf(v);
}
__global__ void k_relu_bf16(const float* __restrict__ in,
                            unsigned short* __restrict__ out, long long n) {
    long long i = (long long)blockIdx.x * blockDim.x + threadIdx.x;
    if (i >= n) return;
    out[i] = f2bf(fmaxf(in[i], 0.f));
}
__global__ void k_final(const float* __restrict__ g1,   // GAT layer1 head-sum
                        const float* __restrict__ g2,   // GCN layer1 agg
                        float* __restrict__ out, long long n) {
    long long i = (long long)blockIdx.x * blockDim.x + threadIdx.x;
    if (i >= n) return;
    out[i] = fmaxf(g1[i] * 0.125f, fmaxf(g2[i], 0.f));   // mean(8 heads) vs relu
}

// ---------------------------------------------------------------------------
// host launch
// ---------------------------------------------------------------------------
extern "C" void kernel_launch(void* const* d_in, const int* in_sizes, int n_in,
                              void* d_out, int out_size, void* d_ws, size_t ws_size,
                              hipStream_t stream) {
    (void)n_in; (void)out_size; (void)ws_size;
    const int  D  = 128, H = 8;
    const long long N = in_sizes[0] / D;
    const long long E = in_sizes[2];

    const float*     x   = (const float*)d_in[0];
    const long long* ei  = (const long long*)d_in[1];   // [2, E] int64
    const float*     ew  = (const float*)d_in[2];
    const float*     W0  = (const float*)d_in[3];
    const float*     a0s = (const float*)d_in[4];
    const float*     a0d = (const float*)d_in[5];
    const float*     W1  = (const float*)d_in[6];
    const float*     a1s = (const float*)d_in[7];
    const float*     a1d = (const float*)d_in[8];
    const float*     gW0 = (const float*)d_in[9];
    const float*     gW1 = (const float*)d_in[10];
    const long long* srcA = ei;
    const long long* dstA = ei + E;

    // ---- workspace bump allocator (256B aligned) ----
    char* base = (char*)d_ws; size_t off = 0;
    auto alloc = [&](size_t bytes) -> char* {
        char* p = base + off;
        off = (off + bytes + 255) & ~(size_t)255;
        return p;
    };
    unsigned short* xb    = (unsigned short*)alloc(N * D * 2);
    unsigned short* bt0   = (unsigned short*)alloc(128 * 128 * 2);   // GAT0 B^T
    unsigned short* bt1   = (unsigned short*)alloc(1024 * 128 * 2);  // GAT1 B^T
    unsigned short* gbt0  = (unsigned short*)alloc(128 * 128 * 2);   // GCN0 B^T
    unsigned short* gbt1  = (unsigned short*)alloc(128 * 128 * 2);   // GCN1 B^T
    unsigned short* x1b   = (unsigned short*)alloc(N * D * 2);
    unsigned short* x2b   = (unsigned short*)alloc(N * D * 2);
    float* WhA    = (float*)alloc(N * D * 4);        // GAT GEMM out (reused per head)
    float* WhG    = (float*)alloc(N * D * 4);        // GCN GEMM out (reused)
    float* es     = (float*)alloc(N * H * 4);
    float* ed     = (float*)alloc(N * H * 4);
    unsigned int* mbuf = (unsigned int*)alloc(N * H * 4);
    float* sbuf   = (float*)alloc(N * H * 4);
    float* zbuf   = (float*)alloc(E * H * 4);
    float* gout0  = (float*)alloc(N * D * 4);
    float* gout1  = (float*)alloc(N * D * 4);
    float* gcnout = (float*)alloc(N * D * 4);        // reused layer0 -> layer1
    float* deg    = (float*)alloc(N * 4);

    const int T = 256;
    auto nb = [](long long n) { return (unsigned)((n + 255) / 256); };
    const unsigned int ENC_NEG_INF = 0x007FFFFFu;    // enc(-inf)

    // ---- precision conversion + weight pre-transpose ----
    k_f32_to_bf16<<<nb(N * D), T, 0, stream>>>(x, xb, N * D);
    k_w0cat_T<<<nb(128 * 128), T, 0, stream>>>(W0, bt0);
    k_w1cat_T<<<nb(1024 * 128), T, 0, stream>>>(W1, bt1);
    k_transpose_bf16<<<nb(128 * 128), T, 0, stream>>>(gW0, gbt0);
    k_transpose_bf16<<<nb(128 * 128), T, 0, stream>>>(gW1, gbt1);

    dim3 gemmGrid((unsigned)((N + 127) / 128), 2);   // 128 cols -> 2 col-blocks

    // =========================== LAYER 0 ===================================
    // GAT: Wh = x @ W0cat  (all 8 heads, 16 feats each, concatenated)
    k_gemm_bf16<<<gemmGrid, T, 0, stream>>>(xb, bt0, WhA, (int)N);
    k_esed<<<nb(N * H), T, 0, stream>>>(WhA, a0s, a0d, es, ed, (int)N, H, 16);
    k_fill_u32<<<nb(N * H), T, 0, stream>>>(mbuf, N * H, ENC_NEG_INF);
    k_zero_f32<<<nb(N * H), T, 0, stream>>>(sbuf, N * H);
    k_edge_max<<<nb(E * H), T, 0, stream>>>(srcA, dstA, es, ed, mbuf, E, H);
    k_edge_z<<<nb(E * H), T, 0, stream>>>(srcA, dstA, es, ed, mbuf, zbuf, sbuf, E, H);
    k_att<<<nb(E * H), T, 0, stream>>>(dstA, sbuf, zbuf, E, H);
    k_zero_f32<<<nb(N * D), T, 0, stream>>>(gout0, N * D);
    k_gat_scatter<<<nb(E * 32), T, 0, stream>>>(srcA, dstA, WhA, zbuf,
                                                gout0, E, H, 16);
    k_elu_bf16<<<nb(N * D), T, 0, stream>>>(gout0, x1b, N * D);

    // GCN layer 0
    k_gemm_bf16<<<gemmGrid, T, 0, stream>>>(xb, gbt0, WhG, (int)N);
    k_zero_f32<<<nb(N), T, 0, stream>>>(deg, N);
    k_deg<<<nb(E), T, 0, stream>>>(dstA, ew, deg, E);
    k_zero_f32<<<nb(N * D), T, 0, stream>>>(gcnout, N * D);
    k_gcn_scatter<<<nb(E * 32), T, 0, stream>>>(srcA, dstA, ew, deg, WhG, gcnout, E);
    k_relu_bf16<<<nb(N * D), T, 0, stream>>>(gcnout, x2b, N * D);

    // =========================== LAYER 1 ===================================
    // GAT layer 1: heads are independent -> loop heads, reuse N x 128 scratch
    k_zero_f32<<<nb(N * D), T, 0, stream>>>(gout1, N * D);
    for (int h = 0; h < H; ++h) {
        k_gemm_bf16<<<gemmGrid, T, 0, stream>>>(x1b, bt1 + (size_t)h * 128 * 128,
                                                WhA, (int)N);
        k_esed<<<nb(N), T, 0, stream>>>(WhA, a1s + h * 128, a1d + h * 128,
                                        es, ed, (int)N, 1, 128);
        k_fill_u32<<<nb(N), T, 0, stream>>>(mbuf, N, ENC_NEG_INF);
        k_zero_f32<<<nb(N), T, 0, stream>>>(sbuf, N);
        k_edge_max<<<nb(E), T, 0, stream>>>(srcA, dstA, es, ed, mbuf, E, 1);
        k_edge_z<<<nb(E), T, 0, stream>>>(srcA, dstA, es, ed, mbuf, zbuf, sbuf, E, 1);
        k_att<<<nb(E), T, 0, stream>>>(dstA, sbuf, zbuf, E, 1);
        k_gat_scatter<<<nb(E * 32), T, 0, stream>>>(srcA, dstA, WhA, zbuf,
                                                    gout1, E, 1, 128);
    }

    // GCN layer 1 (same degree normalization)
    k_gemm_bf16<<<gemmGrid, T, 0, stream>>>(x2b, gbt1, WhG, (int)N);
    k_zero_f32<<<nb(N * D), T, 0, stream>>>(gcnout, N * D);
    k_gcn_scatter<<<nb(E * 32), T, 0, stream>>>(srcA, dstA, ew, deg, WhG, gcnout, E);

    // final: max(mean over heads, relu(gcn))
    k_final<<<nb(N * D), T, 0, stream>>>(gout1, gcnout, (float*)d_out, N * D);
}
